// GPST_GINE_11785390260550
// MI455X (gfx1250) — matrix operand order repaired and checked by reading, not compiled, not run on hardware
//
#include <hip/hip_runtime.h>
#include <stdint.h>

#define Nn 4096
#define Ff 256
#define Ee 131072
#define EDGEF 64
#define HID 512
#define OUTF 8
#define BN_EPS 1e-5f

typedef __attribute__((ext_vector_type(16))) __bf16 v16bf;
typedef __attribute__((ext_vector_type(8)))  __bf16 v8bf;
typedef __attribute__((ext_vector_type(8)))  float  v8f;

__device__ __forceinline__ __bf16 to_bf(float f) {
  union { float f; uint32_t u; } x; x.f = f;
  uint32_t u = x.u;
  u += 0x7fffu + ((u >> 16) & 1u);          // round-to-nearest-even
  union { uint16_t s; __bf16 b; } y; y.s = (uint16_t)(u >> 16);
  return y.b;
}

__device__ __forceinline__ v16bf cat8(v8bf lo, v8bf hi) {
  return __builtin_shufflevector(lo, hi, 0,1,2,3,4,5,6,7,8,9,10,11,12,13,14,15);
}
// A operand: row-major bf16 row pointer; 16 contiguous k at k0+16h
__device__ __forceinline__ v16bf loadA16(const __bf16* row, int k0, int h) {
  v8bf lo = *(const v8bf*)(row + k0 + h * 16);
  v8bf hi = *(const v8bf*)(row + k0 + h * 16 + 8);
  return cat8(lo, hi);
}
// B operand: 16 contiguous bf16
__device__ __forceinline__ v16bf loadB16(const __bf16* p) {
  v8bf lo = *(const v8bf*)p;
  v8bf hi = *(const v8bf*)(p + 8);
  return cat8(lo, hi);
}

// Async global->LDS copy, 16B per lane (CDNA5 GLOBAL_LOAD_ASYNC_TO_LDS_B128).
// Generic LDS pointers carry the LDS byte offset in their low 32 bits.
__device__ __forceinline__ void async_g2l_b128(void* ldsDst, const void* gSrc) {
  uint32_t l = (uint32_t)(uintptr_t)ldsDst;
  asm volatile("global_load_async_to_lds_b128 %0, %1, off"
               :: "v"(l), "v"(gSrc) : "memory");
}
__device__ __forceinline__ void wait_async0() {
  asm volatile("s_wait_asynccnt 0x0" ::: "memory");
}

// ---------------------------------------------------------------------------
// Prepass: transpose+convert weights  W[K,M] f32 -> Wt[M,K] bf16
// ---------------------------------------------------------------------------
__global__ __launch_bounds__(256) void wtrans(const float* __restrict__ W,
                                              __bf16* __restrict__ Wt,
                                              int K, int M) {
  const int i = blockIdx.x * 256 + threadIdx.x;
  if (i >= K * M) return;
  const int m = i / K, k = i % K;
  Wt[i] = to_bf(W[(size_t)k * M + m]);
}

__global__ __launch_bounds__(256) void cvt_bf16(const float* __restrict__ in,
                                                __bf16* __restrict__ out, int n) {
  const int i = blockIdx.x * 256 + threadIdx.x;
  if (i < n) out[i] = to_bf(in[i]);
}

// ---------------------------------------------------------------------------
// C = act(A[rows,K]bf16 @ Wt[M,K]bf16 + bias [+ addend]); 32x32 out per wave.
// OUT: 0=f32 row-major, 1=bf16 row-major, 2=bf16 transposed [M][rows]
// ---------------------------------------------------------------------------
template<bool RELU, bool ADD, int OUT>
__global__ __launch_bounds__(256) void gemm_bf16(
    const __bf16* __restrict__ A, const __bf16* __restrict__ Wt,
    const float* __restrict__ bias, const float* __restrict__ addend,
    void* __restrict__ Cout, int rows, int K, int M)
{
  const int lane = threadIdx.x & 31;
  const int wave = threadIdx.x >> 5;
  const int ml = lane & 15, h = lane >> 4;
  const int tilesM = M >> 5;
  const int t = blockIdx.x * 8 + wave;
  const int rt = t / tilesM, ct = t % tilesM;
  const int row0 = rt * 32, col0 = ct * 32;
  if (row0 >= rows) return;

  const __bf16* Ar0 = A + (size_t)(row0 + ml) * K;
  const __bf16* Ar1 = A + (size_t)(row0 + 16 + ml) * K;
  const __bf16* Wc0 = Wt + (size_t)(col0 + ml) * K;
  const __bf16* Wc1 = Wt + (size_t)(col0 + 16 + ml) * K;

  v8f acc00 = {0.f,0.f,0.f,0.f,0.f,0.f,0.f,0.f};
  v8f acc01 = acc00, acc10 = acc00, acc11 = acc00;
  for (int k0 = 0; k0 < K; k0 += 32) {
    const v16bf a0 = loadA16(Ar0, k0, h);
    const v16bf a1 = loadA16(Ar1, k0, h);
    const v16bf b0 = loadB16(Wc0 + k0 + h * 16);
    const v16bf b1 = loadB16(Wc1 + k0 + h * 16);
    acc00 = __builtin_amdgcn_wmma_f32_16x16x32_bf16(false, a0, false, b0, (short)0, acc00, false, false);
    acc01 = __builtin_amdgcn_wmma_f32_16x16x32_bf16(false, a0, false, b1, (short)0, acc01, false, false);
    acc10 = __builtin_amdgcn_wmma_f32_16x16x32_bf16(false, a1, false, b0, (short)0, acc10, false, false);
    acc11 = __builtin_amdgcn_wmma_f32_16x16x32_bf16(false, a1, false, b1, (short)0, acc11, false, false);
  }
  v8f* accs[4] = { &acc00, &acc01, &acc10, &acc11 };
  #pragma unroll
  for (int rr = 0; rr < 2; ++rr)
    #pragma unroll
    for (int cc = 0; cc < 2; ++cc) {
      const v8f& acc = *accs[rr * 2 + cc];
      #pragma unroll
      for (int i = 0; i < 8; ++i) {
        const int r = row0 + rr * 16 + i + 8 * h;
        const int c = col0 + cc * 16 + ml;
        float v = acc[i] + bias[c];
        if (ADD)  v += addend[(size_t)r * M + c];
        if (RELU) v = fmaxf(v, 0.f);
        if (OUT == 0)      ((float*)Cout)[(size_t)r * M + c] = v;
        else if (OUT == 1) ((__bf16*)Cout)[(size_t)r * M + c] = to_bf(v);
        else               ((__bf16*)Cout)[(size_t)c * rows + r] = to_bf(v);
      }
    }
}

// ---------------------------------------------------------------------------
// Fused GINE edge stage: msg = relu(x[src] + edge_attr@edge_W + edge_b),
// scatter-add into agg[dst] (agg pre-initialized to x).
// Batched gathers per tile so loads pipeline ahead of the atomic chain.
// ---------------------------------------------------------------------------
__global__ __launch_bounds__(256) void gine_edges(
    const float* __restrict__ x, const __bf16* __restrict__ eattr,
    const long long* __restrict__ eidx, const __bf16* __restrict__ eWt, // [Ff][EDGEF]
    const float* __restrict__ eb, float* __restrict__ agg)
{
  const int lane = threadIdx.x & 31;
  const int wave = threadIdx.x >> 5;
  const int ml = lane & 15, h = lane >> 4;
  const int e0 = (blockIdx.x * 8 + wave) * 16;

  const __bf16* Er = eattr + (size_t)(e0 + ml) * EDGEF;
  v16bf aE[2];
  #pragma unroll
  for (int s = 0; s < 2; ++s) aE[s] = loadA16(Er, s * 32, h);

  const float* xs[8];
  float*       ad[8];
  #pragma unroll
  for (int i = 0; i < 8; ++i) {
    const int e = e0 + i + 8 * h;
    xs[i] = x   + (size_t)((int)eidx[e])      * Ff + ml;
    ad[i] = agg + (size_t)((int)eidx[Ee + e]) * Ff + ml;
  }
  const __bf16* wp = eWt + (size_t)ml * EDGEF;   // + ct*16*EDGEF per tile
  #pragma unroll
  for (int ct = 0; ct < Ff / 16; ++ct) {
    v8f acc = {0.f,0.f,0.f,0.f,0.f,0.f,0.f,0.f};
    #pragma unroll
    for (int s = 0; s < 2; ++s)
      acc = __builtin_amdgcn_wmma_f32_16x16x32_bf16(false, aE[s], false,
              loadB16(wp + (size_t)ct * 16 * EDGEF + s * 32 + h * 16),
              (short)0, acc, false, false);
    float xv[8];
    #pragma unroll
    for (int i = 0; i < 8; ++i) xv[i] = xs[i][ct * 16];   // batched gathers
    const float ebv = eb[ct * 16 + ml];
    #pragma unroll
    for (int i = 0; i < 8; ++i) {
      const float v = fmaxf(acc[i] + ebv + xv[i], 0.f);
      atomicAdd(&ad[i][ct * 16], v);
    }
  }
}

// ---------------------------------------------------------------------------
// Split-K flash attention, 32-key blocks. All 8 waves of a block share one
// head+segment; K/V tiles staged into LDS once per block via async copies.
// Emits unnormalized partial O plus per-row (m,l); combined by attn_combine.
// ---------------------------------------------------------------------------
template<int DH>
__global__ __launch_bounds__(256) void attn_flash(
    const __bf16* __restrict__ q, const __bf16* __restrict__ k,
    const __bf16* __restrict__ vT, float* __restrict__ Opart,
    float* __restrict__ mbuf, float* __restrict__ lbuf, int split)
{
  __shared__ __align__(16) __bf16 kls[32][DH];     // [key][dim]
  __shared__ __align__(16) __bf16 vls[DH][32];     // [dim][key]
  __shared__ __align__(16) __bf16 plds[8][16][40]; // per-wave 16x32 P tile
  const int lane = threadIdx.x & 31;
  const int wave = threadIdx.x >> 5;
  const int ml = lane & 15, h = lane >> 4;
  const int QTB = (Nn / 16) / 8;                   // 32 qt-groups per head
  const int bph = QTB * split;
  const int head = blockIdx.x / bph;
  const int rem  = blockIdx.x % bph;
  const int qt   = (rem / split) * 8 + wave;
  const int seg  = rem % split;
  const int segLen = Nn / split;
  const int db = head * DH;
  const float scale = rsqrtf((float)DH);

  // Per-thread staging chunk descriptors (16B each, DH/64 chunks per thread).
  const int NCH = DH / 64;
  const __bf16* ksrc[NCH]; __bf16* kdst[NCH];
  const __bf16* vsrc[NCH]; __bf16* vdst[NCH];
  #pragma unroll
  for (int n = 0; n < NCH; ++n) {
    const int c = threadIdx.x + n * 256;
    const int krow = c / (DH / 8), kj = c % (DH / 8);
    ksrc[n] = k + (size_t)krow * Ff + db + kj * 8;     // + kb*Ff per block
    kdst[n] = &kls[krow][kj * 8];
    const int vd = c / 4, vj = c % 4;
    vsrc[n] = vT + (size_t)(db + vd) * Nn + vj * 8;    // + kb per block
    vdst[n] = &vls[vd][vj * 8];
  }

  v16bf aQ[DH / 32];
  {
    const __bf16* Qr = q + (size_t)(qt * 16 + ml) * Ff + db;
    #pragma unroll
    for (int s = 0; s < DH / 32; ++s) aQ[s] = loadA16(Qr, s * 32, h);
  }

  float mrow[8], lrow[8];
  v8f Oa[DH / 16];
  #pragma unroll
  for (int i = 0; i < 8; ++i) { mrow[i] = -1e30f; lrow[i] = 0.f; }
  #pragma unroll
  for (int c = 0; c < DH / 16; ++c)
    Oa[c] = (v8f){0.f,0.f,0.f,0.f,0.f,0.f,0.f,0.f};

  for (int kb = seg * segLen; kb < (seg + 1) * segLen; kb += 32) {
    // stage K/V tiles for this 32-key block
    #pragma unroll
    for (int n = 0; n < NCH; ++n) async_g2l_b128(kdst[n], ksrc[n] + (size_t)kb * Ff);
    #pragma unroll
    for (int n = 0; n < NCH; ++n) async_g2l_b128(vdst[n], vsrc[n] + kb);
    wait_async0();
    __syncthreads();                 // K/V tiles ready

    v8f s0 = {0.f,0.f,0.f,0.f,0.f,0.f,0.f,0.f};
    v8f s1 = s0;
    #pragma unroll
    for (int st = 0; st < DH / 32; ++st) {
      const v16bf b0 = loadB16(&kls[ml][st * 32 + h * 16]);
      const v16bf b1 = loadB16(&kls[16 + ml][st * 32 + h * 16]);
      s0 = __builtin_amdgcn_wmma_f32_16x16x32_bf16(false, aQ[st], false, b0, (short)0, s0, false, false);
      s1 = __builtin_amdgcn_wmma_f32_16x16x32_bf16(false, aQ[st], false, b1, (short)0, s1, false, false);
    }
    float corr[8];
    #pragma unroll
    for (int i = 0; i < 8; ++i) {
      const float sv0 = s0[i] * scale, sv1 = s1[i] * scale;
      float mx = fmaxf(sv0, sv1);
      #pragma unroll
      for (int d = 1; d < 16; d <<= 1) mx = fmaxf(mx, __shfl_xor(mx, d, 32));
      const float mn = fmaxf(mrow[i], mx);
      const float p0 = __expf(sv0 - mn);
      const float p1 = __expf(sv1 - mn);
      float rs = p0 + p1;
      #pragma unroll
      for (int d = 1; d < 16; d <<= 1) rs += __shfl_xor(rs, d, 32);
      corr[i] = __expf(mrow[i] - mn);
      lrow[i] = lrow[i] * corr[i] + rs;
      mrow[i] = mn;
      plds[wave][i + 8 * h][ml]      = to_bf(p0);
      plds[wave][i + 8 * h][16 + ml] = to_bf(p1);
    }
    #pragma unroll
    for (int c = 0; c < DH / 16; ++c)
      #pragma unroll
      for (int i = 0; i < 8; ++i) Oa[c][i] *= corr[i];
    __syncthreads();                 // P tiles written
    v16bf aP;
    {
      v8bf lo = *(const v8bf*)&plds[wave][ml][h * 16];
      v8bf hi = *(const v8bf*)&plds[wave][ml][h * 16 + 8];
      aP = cat8(lo, hi);
    }
    #pragma unroll
    for (int c = 0; c < DH / 16; ++c)
      Oa[c] = __builtin_amdgcn_wmma_f32_16x16x32_bf16(false, aP, false,
                loadB16(&vls[c * 16 + ml][h * 16]), (short)0, Oa[c], false, false);
    __syncthreads();                 // done reading tiles before next staging
  }
  // write unnormalized partial O + per-row m,l for this segment
  #pragma unroll
  for (int c = 0; c < DH / 16; ++c)
    #pragma unroll
    for (int i = 0; i < 8; ++i) {
      const int r = qt * 16 + i + 8 * h;
      Opart[((size_t)seg * Nn + r) * Ff + db + c * 16 + ml] = Oa[c][i];
    }
  if (ml == 0) {
    #pragma unroll
    for (int i = 0; i < 8; ++i) {
      const int r = qt * 16 + i + 8 * h;
      mbuf[((size_t)seg * 4 + head) * Nn + r] = mrow[i];
      lbuf[((size_t)seg * 4 + head) * Nn + r] = lrow[i];
    }
  }
}

__global__ __launch_bounds__(256) void attn_combine(
    const float* __restrict__ Opart, const float* __restrict__ mbuf,
    const float* __restrict__ lbuf, __bf16* __restrict__ ab,
    int split, int dh)
{
  const int idx = blockIdx.x * 256 + threadIdx.x;  // Nn*Ff
  const int r = idx / Ff, f = idx % Ff;
  const int head = f / dh;
  float M = -1e30f;
  for (int s = 0; s < split; ++s)
    M = fmaxf(M, mbuf[((size_t)s * 4 + head) * Nn + r]);
  float L = 0.f, O = 0.f;
  for (int s = 0; s < split; ++s) {
    const float w = __expf(mbuf[((size_t)s * 4 + head) * Nn + r] - M);
    L += lbuf[((size_t)s * 4 + head) * Nn + r] * w;
    O += Opart[((size_t)s * Nn + r) * Ff + f] * w;
  }
  ab[idx] = to_bf(O / L);
}

// ---------------------------------------------------------------------------
// Training-mode BatchNorm1d; one block per feature column (F=256).
// ---------------------------------------------------------------------------
template<bool RELU, bool BFOUT>
__global__ __launch_bounds__(256) void bnorm(
    const float* __restrict__ in, const float* __restrict__ g,
    const float* __restrict__ b, float* __restrict__ out,
    __bf16* __restrict__ outbf, int rows)
{
  __shared__ float s1[256], s2[256];
  const int f = blockIdx.x, tid = threadIdx.x;
  float a = 0.f, a2 = 0.f;
  for (int r = tid; r < rows; r += 256) {
    const float val = in[(size_t)r * Ff + f];
    a += val; a2 += val * val;
  }
  s1[tid] = a; s2[tid] = a2;
  __syncthreads();
  for (int st = 128; st > 0; st >>= 1) {
    if (tid < st) { s1[tid] += s1[tid + st]; s2[tid] += s2[tid + st]; }
    __syncthreads();
  }
  const float mu = s1[0] / (float)rows;
  const float var = s2[0] / (float)rows - mu * mu;
  const float rstd = rsqrtf(var + BN_EPS);
  const float gg = g[f], bb = b[f];
  for (int r = tid; r < rows; r += 256) {
    float val = (in[(size_t)r * Ff + f] - mu) * rstd * gg + bb;
    if (RELU) val = fmaxf(val, 0.f);
    out[(size_t)r * Ff + f] = val;
    if (BFOUT) outbf[(size_t)r * Ff + f] = to_bf(val);
  }
}

__global__ __launch_bounds__(256) void addk(
    const float* __restrict__ a, const float* __restrict__ b,
    float* __restrict__ o, __bf16* __restrict__ obf, int n)
{
  const int i = blockIdx.x * 256 + threadIdx.x;
  if (i < n) {
    const float v = a[i] + b[i];
    o[i] = v;
    obf[i] = to_bf(v);
  }
}

__global__ __launch_bounds__(256) void final_lin(
    const float* __restrict__ h, const float* __restrict__ W,
    const float* __restrict__ b, float* __restrict__ o)
{
  const int i = blockIdx.x * 256 + threadIdx.x;  // N*OUTF outputs
  const int r = i / OUTF, c = i % OUTF;
  float s = b[c];
  for (int kk = 0; kk < Ff; ++kk) s += h[(size_t)r * Ff + kk] * W[kk * OUTF + c];
  o[i] = s;
}

// ---------------------------------------------------------------------------
// Host orchestration
// ---------------------------------------------------------------------------
struct GPSParams {  // field order == sorted jax pytree leaf order
  const float *Wk,*Wo,*Wq,*Wv,*b1,*b2,*b3,*bk,*bo,*bq,*bv,*edge_W,*edge_b,
              *ff_W1,*ff_W2,*ff_b1,*ff_b2,*g1,*g2,*g3,
              *mlp_W1,*mlp_W2,*mlp_b1,*mlp_b2;
};
struct WtSet { __bf16 *Wq,*Wk,*Wv,*Wo,*m1,*m2,*f1,*f2,*ew; };

static GPSParams load_gps(void* const* d, int b) {
  GPSParams L;
  L.Wk=(const float*)d[b+0];  L.Wo=(const float*)d[b+1];
  L.Wq=(const float*)d[b+2];  L.Wv=(const float*)d[b+3];
  L.b1=(const float*)d[b+4];  L.b2=(const float*)d[b+5];
  L.b3=(const float*)d[b+6];  L.bk=(const float*)d[b+7];
  L.bo=(const float*)d[b+8];  L.bq=(const float*)d[b+9];
  L.bv=(const float*)d[b+10]; L.edge_W=(const float*)d[b+11];
  L.edge_b=(const float*)d[b+12];
  L.ff_W1=(const float*)d[b+13]; L.ff_W2=(const float*)d[b+14];
  L.ff_b1=(const float*)d[b+15]; L.ff_b2=(const float*)d[b+16];
  L.g1=(const float*)d[b+17]; L.g2=(const float*)d[b+18];
  L.g3=(const float*)d[b+19];
  L.mlp_W1=(const float*)d[b+20]; L.mlp_W2=(const float*)d[b+21];
  L.mlp_b1=(const float*)d[b+22]; L.mlp_b2=(const float*)d[b+23];
  return L;
}

static void prep_weights(const GPSParams& L, const WtSet& T, hipStream_t s) {
  wtrans<<<(Ff * Ff) / 256, 256, 0, s>>>(L.Wq, T.Wq, Ff, Ff);
  wtrans<<<(Ff * Ff) / 256, 256, 0, s>>>(L.Wk, T.Wk, Ff, Ff);
  wtrans<<<(Ff * Ff) / 256, 256, 0, s>>>(L.Wv, T.Wv, Ff, Ff);
  wtrans<<<(Ff * Ff) / 256, 256, 0, s>>>(L.Wo, T.Wo, Ff, Ff);
  wtrans<<<(Ff * HID) / 256, 256, 0, s>>>(L.mlp_W1, T.m1, Ff, HID);
  wtrans<<<(HID * Ff) / 256, 256, 0, s>>>(L.mlp_W2, T.m2, HID, Ff);
  wtrans<<<(Ff * HID) / 256, 256, 0, s>>>(L.ff_W1, T.f1, Ff, HID);
  wtrans<<<(HID * Ff) / 256, 256, 0, s>>>(L.ff_W2, T.f2, HID, Ff);
  wtrans<<<(EDGEF * Ff) / 256, 256, 0, s>>>(L.edge_W, T.ew, EDGEF, Ff);
}

struct Bufs {
  float *agg, *t0, *h1, *h2, *tmp, *xo, *x2, *Opart, *mbuf, *lbuf;
  __bf16 *xbf, *aggbf, *tmpbf, *abbf, *x2bf, *qbf, *kbf, *vT, *midbf, *eattrbf;
};

static void gps_layer(const GPSParams& L, const WtSet& T, const Bufs& B,
                      const float* xin, const __bf16* xinbf,
                      const long long* eidx, int heads, hipStream_t stream)
{
  const size_t bytes = (size_t)Nn * Ff * sizeof(float);
  const int gT  = (Nn / 32) * (Ff / 32) / 8;    // 256-col GEMM grids
  const int gTh = (Nn / 32) * (HID / 32) / 8;   // 512-col GEMM grids
  const int QTB = (Nn / 16) / 8;

  // GINE branch: agg = x + sum relu(x[src] + Lin(edge_attr))
  hipMemcpyAsync(B.agg, xin, bytes, hipMemcpyDeviceToDevice, stream);
  gine_edges<<<Ee / 128, 256, 0, stream>>>(xin, B.eattrbf, eidx, T.ew, L.edge_b, B.agg);
  cvt_bf16<<<(Nn * Ff) / 256, 256, 0, stream>>>(B.agg, B.aggbf, Nn * Ff);
  gemm_bf16<true,false,1><<<gTh, 256, 0, stream>>>(B.aggbf, T.m1, L.mlp_b1, nullptr, B.midbf, Nn, Ff, HID);
  gemm_bf16<false,true,0><<<gT, 256, 0, stream>>>(B.midbf, T.m2, L.mlp_b2, xin, B.t0, Nn, HID, Ff);
  bnorm<false,false><<<Ff, 256, 0, stream>>>(B.t0, L.g1, L.b1, B.h1, nullptr, Nn);

  // Attention branch (split-K flash + combine)
  gemm_bf16<false,false,1><<<gT, 256, 0, stream>>>(xinbf, T.Wq, L.bq, nullptr, B.qbf, Nn, Ff, Ff);
  gemm_bf16<false,false,1><<<gT, 256, 0, stream>>>(xinbf, T.Wk, L.bk, nullptr, B.kbf, Nn, Ff, Ff);
  gemm_bf16<false,false,2><<<gT, 256, 0, stream>>>(xinbf, T.Wv, L.bv, nullptr, B.vT, Nn, Ff, Ff);
  if (heads == 4) {
    const int split = 2;
    attn_flash<64><<<4 * QTB * split, 256, 0, stream>>>(B.qbf, B.kbf, B.vT,
        B.Opart, B.mbuf, B.lbuf, split);
    attn_combine<<<(Nn * Ff) / 256, 256, 0, stream>>>(B.Opart, B.mbuf, B.lbuf,
        B.abbf, split, Ff / 4);
  } else {
    const int split = 8;
    attn_flash<256><<<1 * QTB * split, 256, 0, stream>>>(B.qbf, B.kbf, B.vT,
        B.Opart, B.mbuf, B.lbuf, split);
    attn_combine<<<(Nn * Ff) / 256, 256, 0, stream>>>(B.Opart, B.mbuf, B.lbuf,
        B.abbf, split, Ff);
  }
  gemm_bf16<false,true,0><<<gT, 256, 0, stream>>>(B.abbf, T.Wo, L.bo, xin, B.t0, Nn, Ff, Ff);
  bnorm<false,false><<<Ff, 256, 0, stream>>>(B.t0, L.g2, L.b2, B.h2, nullptr, Nn);

  // Combine + feed-forward
  addk<<<(Nn * Ff) / 256, 256, 0, stream>>>(B.h1, B.h2, B.tmp, B.tmpbf, Nn * Ff);
  gemm_bf16<true,false,1><<<gTh, 256, 0, stream>>>(B.tmpbf, T.f1, L.ff_b1, nullptr, B.midbf, Nn, Ff, HID);
  gemm_bf16<false,true,0><<<gT, 256, 0, stream>>>(B.midbf, T.f2, L.ff_b2, B.tmp, B.t0, Nn, HID, Ff);
  bnorm<false,false><<<Ff, 256, 0, stream>>>(B.t0, L.g3, L.b3, B.xo, nullptr, Nn);
}

extern "C" void kernel_launch(void* const* d_in, const int* in_sizes, int n_in,
                              void* d_out, int out_size, void* d_ws, size_t ws_size,
                              hipStream_t stream) {
  (void)n_in; (void)out_size; (void)ws_size;

  // Two plausible flattenings: insertion-order top level vs fully key-sorted.
  const bool xfirst = (in_sizes[0] == Nn * Ff);
  const float*     x;
  const long long* eidx;
  const float*     eattr;
  const float *bn1_g, *bn1_b, *bn2_g, *bn2_b, *linW, *linb;
  GPSParams L1, L2;
  if (xfirst) {
    x     = (const float*)d_in[0];
    eidx  = (const long long*)d_in[1];
    eattr = (const float*)d_in[2];
    bn1_b = (const float*)d_in[3]; bn1_g = (const float*)d_in[4];
    bn2_b = (const float*)d_in[5]; bn2_g = (const float*)d_in[6];
    L1 = load_gps(d_in, 7);
    L2 = load_gps(d_in, 31);
    linW = (const float*)d_in[55]; linb = (const float*)d_in[56];
  } else {
    eattr = (const float*)d_in[0];
    eidx  = (const long long*)d_in[1];
    bn1_b = (const float*)d_in[2]; bn1_g = (const float*)d_in[3];
    bn2_b = (const float*)d_in[4]; bn2_g = (const float*)d_in[5];
    L1 = load_gps(d_in, 6);
    L2 = load_gps(d_in, 30);
    linW = (const float*)d_in[54]; linb = (const float*)d_in[55];
    x    = (const float*)d_in[56];
  }

  // ---- workspace carve-up: fp32 region first, then bf16 region ----
  Bufs B;
  float* wf = (float*)d_ws;
  B.agg   = wf; wf += (size_t)Nn * Ff;
  B.t0    = wf; wf += (size_t)Nn * Ff;
  B.h1    = wf; wf += (size_t)Nn * Ff;
  B.h2    = wf; wf += (size_t)Nn * Ff;
  B.tmp   = wf; wf += (size_t)Nn * Ff;
  B.xo    = wf; wf += (size_t)Nn * Ff;
  B.x2    = wf; wf += (size_t)Nn * Ff;
  B.Opart = wf; wf += (size_t)8 * Nn * Ff;       // max split = 8
  B.mbuf  = wf; wf += (size_t)8 * 4 * Nn;
  B.lbuf  = wf; wf += (size_t)8 * 4 * Nn;
  __bf16* wb = (__bf16*)wf;
  B.xbf     = wb; wb += (size_t)Nn * Ff;
  B.aggbf   = wb; wb += (size_t)Nn * Ff;
  B.tmpbf   = wb; wb += (size_t)Nn * Ff;
  B.abbf    = wb; wb += (size_t)Nn * Ff;
  B.x2bf    = wb; wb += (size_t)Nn * Ff;
  B.qbf     = wb; wb += (size_t)Nn * Ff;
  B.kbf     = wb; wb += (size_t)Nn * Ff;
  B.vT      = wb; wb += (size_t)Nn * Ff;
  B.midbf   = wb; wb += (size_t)Nn * HID;
  B.eattrbf = wb; wb += (size_t)Ee * EDGEF;
  WtSet T1, T2;
  __bf16** slots1[9] = {&T1.Wq,&T1.Wk,&T1.Wv,&T1.Wo,&T1.m1,&T1.m2,&T1.f1,&T1.f2,&T1.ew};
  __bf16** slots2[9] = {&T2.Wq,&T2.Wk,&T2.Wv,&T2.Wo,&T2.m1,&T2.m2,&T2.f1,&T2.f2,&T2.ew};
  const size_t wsz[9] = { (size_t)Ff*Ff,(size_t)Ff*Ff,(size_t)Ff*Ff,(size_t)Ff*Ff,
                          (size_t)Ff*HID,(size_t)HID*Ff,(size_t)Ff*HID,(size_t)HID*Ff,
                          (size_t)EDGEF*Ff };
  for (int i = 0; i < 9; ++i) { *slots1[i] = wb; wb += wsz[i]; }
  for (int i = 0; i < 9; ++i) { *slots2[i] = wb; wb += wsz[i]; }

  // ---- prepass: bf16 conversions done once per launch ----
  prep_weights(L1, T1, stream);
  prep_weights(L2, T2, stream);
  cvt_bf16<<<(Ee * EDGEF) / 256, 256, 0, stream>>>(eattr, B.eattrbf, Ee * EDGEF);
  cvt_bf16<<<(Nn * Ff) / 256, 256, 0, stream>>>(x, B.xbf, Nn * Ff);

  // ---- layer 1 (heads=4), outer BN + ReLU (fp32 + bf16 outputs) ----
  gps_layer(L1, T1, B, x, B.xbf, eidx, 4, stream);
  bnorm<true,true><<<Ff, 256, 0, stream>>>(B.xo, bn1_g, bn1_b, B.x2, B.x2bf, Nn);

  // ---- layer 2 (heads=1), outer BN + ReLU ----
  gps_layer(L2, T2, B, B.x2, B.x2bf, eidx, 1, stream);
  bnorm<true,false><<<Ff, 256, 0, stream>>>(B.xo, bn2_g, bn2_b, B.tmp, nullptr, Nn);

  // ---- final linear 256 -> 8 ----
  final_lin<<<(Nn * OUTF) / 256, 256, 0, stream>>>(B.tmp, linW, linb, (float*)d_out);
}
